// DbrxAttention_44186623541248
// MI455X (gfx1250) — compile-verified, matrix operand order checked by reading, not run on hardware
//
#include <hip/hip_runtime.h>

#define D_MODEL   4096
#define N_HEADS   32
#define KV_HEADS  8
#define HEAD_DIM  128
#define T_TOK     2048
#define QKV_N     (D_MODEL + 2 * KV_HEADS * HEAD_DIM)   /* 6144 */
#define QKV_LD    QKV_N

#define USE_TDM   1

typedef __attribute__((ext_vector_type(16))) __bf16 v16bf;
typedef __attribute__((ext_vector_type(8)))  float  v8f;
typedef __attribute__((ext_vector_type(4)))  unsigned int u32x4;
typedef __attribute__((ext_vector_type(8)))  unsigned int u32x8;

union Frag16 { v16bf v; uint4 q[2]; };

__device__ __forceinline__ unsigned short f32_to_bf16(float f) {
  unsigned int u = __float_as_uint(f);
  u += 0x7FFFu + ((u >> 16) & 1u);            // round-to-nearest-even
  return (unsigned short)(u >> 16);
}

__device__ __forceinline__ uint4 pack_bf16x8_f(const float* p) {
  uint4 o;
  o.x = (unsigned)f32_to_bf16(p[0]) | ((unsigned)f32_to_bf16(p[1]) << 16);
  o.y = (unsigned)f32_to_bf16(p[2]) | ((unsigned)f32_to_bf16(p[3]) << 16);
  o.z = (unsigned)f32_to_bf16(p[4]) | ((unsigned)f32_to_bf16(p[5]) << 16);
  o.w = (unsigned)f32_to_bf16(p[6]) | ((unsigned)f32_to_bf16(p[7]) << 16);
  return o;
}

__device__ __forceinline__ uint4 pack_u16x8(const unsigned short* t) {
  uint4 o;
  o.x = (unsigned)t[0] | ((unsigned)t[1] << 16);
  o.y = (unsigned)t[2] | ((unsigned)t[3] << 16);
  o.z = (unsigned)t[4] | ((unsigned)t[5] << 16);
  o.w = (unsigned)t[6] | ((unsigned)t[7] << 16);
  return o;
}

#if USE_TDM
// Issue a TDM DMA of a 128-row x 32-elem bf16 tile (row stride = K elems,
// rows packed contiguously in LDS) from global to LDS. Wave-level op,
// tracked with TENSORcnt. 2-operand (<=2D) form of tensor_load_to_lds.
__device__ __forceinline__ void tdm_load_tile_128x32(
    const unsigned short* gsrc, const unsigned short* lds_dst, unsigned k_elems) {
  unsigned long long ga = (unsigned long long)gsrc;
  unsigned lds_off = (unsigned)(unsigned long long)lds_dst;  // flat addr low 32b = LDS offset
  u32x4 g0;
  g0[0] = 1u;                                        // count=1, user mode
  g0[1] = lds_off;                                   // lds_addr (bytes)
  g0[2] = (unsigned)(ga & 0xFFFFFFFFull);            // global_addr[31:0]
  g0[3] = (unsigned)((ga >> 32) & 0x01FFFFFFull)     // global_addr[56:32]
          | (2u << 30);                              // type=2 ("image")
  u32x8 g1;
  g1[0] = 1u << 16;                                  // wg_mask=0, data_size=1 (2B)
  g1[1] = 32u << 16;                                 // tensor_dim0[15:0]=32 (bits 63:48)
  g1[2] = 128u << 16;                                // tensor_dim1[15:0]=128 (bits 111:96->95:80)
  g1[3] = 32u << 16;                                 // tile_dim0=32 (bits 127:112)
  g1[4] = 128u;                                      // tile_dim1=128 (bits 143:128)
  g1[5] = k_elems;                                   // tensor_dim0_stride[31:0] (bits 191:160)
  g1[6] = 0u;                                        // stride hi / dim1_stride lo
  g1[7] = 0u;
  asm volatile("tensor_load_to_lds %0, %1" :: "s"(g0), "s"(g1) : "memory");
}
#endif

// ---------------------------------------------------------------- fp32->bf16
__global__ void cvt_bf16_kernel(const float* __restrict__ in,
                                unsigned short* __restrict__ out, int n) {
  int i = blockIdx.x * blockDim.x + threadIdx.x;
  if (i < n) out[i] = f32_to_bf16(in[i]);
}

// ------------------------------------------------------ C = A * B^T (bf16)
// A: M x K row-major bf16; B: N x K row-major bf16; C: M x N fp32.
// 128x128x32 macro-tile, 8 waves (2x4), each wave 64x32 via 8 WMMA tiles.
// Operand staging: double-buffered Tensor Data Mover DMA (global -> LDS).
__global__ __launch_bounds__(256) void gemm_nt_bf16_kernel(
    const unsigned short* __restrict__ A, const unsigned short* __restrict__ B,
    float* __restrict__ C, int M, int N, int K, float clampv) {
  __shared__ __align__(16) unsigned short As[2][128 * 32];
  __shared__ __align__(16) unsigned short Bs[2][128 * 32];
  const int tid  = threadIdx.x;
  const int row0 = blockIdx.y * 128;
  const int col0 = blockIdx.x * 128;
  const int wid  = tid >> 5;
  const int lane = tid & 31;
  const int r    = lane & 15;
  const int half = lane >> 4;
  const int waveRow = wid >> 2;   // 0..1 -> 64 rows each
  const int waveCol = wid & 3;    // 0..3 -> 32 cols each

  v8f acc[4][2];
#pragma unroll
  for (int mt = 0; mt < 4; ++mt)
#pragma unroll
    for (int nt = 0; nt < 2; ++nt)
#pragma unroll
      for (int j = 0; j < 8; ++j) acc[mt][nt][j] = 0.0f;

#if USE_TDM
  if (tid < 32) {  // wave 0 drives the DMA engine for the whole workgroup
    tdm_load_tile_128x32(A + (size_t)row0 * K, &As[0][0], (unsigned)K);
    tdm_load_tile_128x32(B + (size_t)col0 * K, &Bs[0][0], (unsigned)K);
  }
#endif

  int cur = 0;
  for (int k0 = 0; k0 < K; k0 += 32) {
#if USE_TDM
    if (tid < 32) __builtin_amdgcn_s_wait_tensorcnt(0);
    __syncthreads();                       // tile(cur) visible to all waves
    if (k0 + 32 < K && tid < 32) {         // prefetch next tiles into other buffer
      tdm_load_tile_128x32(A + (size_t)row0 * K + (k0 + 32), &As[cur ^ 1][0], (unsigned)K);
      tdm_load_tile_128x32(B + (size_t)col0 * K + (k0 + 32), &Bs[cur ^ 1][0], (unsigned)K);
    }
#else
#pragma unroll
    for (int p = 0; p < 4; ++p) {          // 256 thr * 4 * uint2 = 128x32
      int rr = p * 32 + (tid >> 3);
      int cc = (tid & 7) * 4;
      *(uint2*)&As[0][rr * 32 + cc] =
          *(const uint2*)(A + (size_t)(row0 + rr) * K + k0 + cc);
      *(uint2*)&Bs[0][rr * 32 + cc] =
          *(const uint2*)(B + (size_t)(col0 + rr) * K + k0 + cc);
    }
    __syncthreads();
#endif

    Frag16 fa[4], fb[2];
#pragma unroll
    for (int mt = 0; mt < 4; ++mt) {
      int m = waveRow * 64 + mt * 16 + r;
      fa[mt].q[0] = *(const uint4*)&As[cur][m * 32 + half * 8];
      fa[mt].q[1] = *(const uint4*)&As[cur][m * 32 + 16 + half * 8];
    }
#pragma unroll
    for (int nt = 0; nt < 2; ++nt) {
      int n = waveCol * 32 + nt * 16 + r;
      fb[nt].q[0] = *(const uint4*)&Bs[cur][n * 32 + half * 8];
      fb[nt].q[1] = *(const uint4*)&Bs[cur][n * 32 + 16 + half * 8];
    }
#pragma unroll
    for (int mt = 0; mt < 4; ++mt)
#pragma unroll
      for (int nt = 0; nt < 2; ++nt)
        acc[mt][nt] = __builtin_amdgcn_wmma_f32_16x16x32_bf16(
            false, fa[mt].v, false, fb[nt].v, (short)0, acc[mt][nt], false, false);
    __syncthreads();
#if USE_TDM
    cur ^= 1;
#endif
  }

#pragma unroll
  for (int mt = 0; mt < 4; ++mt)
#pragma unroll
    for (int nt = 0; nt < 2; ++nt)
#pragma unroll
      for (int j = 0; j < 8; ++j) {
        int m = row0 + waveRow * 64 + mt * 16 + j + 8 * half;
        int n = col0 + waveCol * 32 + nt * 16 + r;
        float v = acc[mt][nt][j];
        if (clampv > 0.0f) v = fminf(fmaxf(v, -clampv), clampv);
        C[(size_t)m * N + n] = v;
      }
}

// ------------------------------------------------------------------- RoPE
// In-place on fp32 qkv buffer: 32 q heads + 8 k heads, pairs (d, d+64).
__global__ void rope_kernel(float* __restrict__ qkv,
                            const float* __restrict__ cs,
                            const float* __restrict__ sn) {
  int t  = blockIdx.x;
  int hh = blockIdx.y;          // 0..39
  int d  = threadIdx.x;         // 0..63
  float c = cs[t * 64 + d];
  float s = sn[t * 64 + d];
  float* base = (hh < N_HEADS)
      ? (qkv + (size_t)t * QKV_LD + hh * HEAD_DIM)
      : (qkv + (size_t)t * QKV_LD + D_MODEL + (hh - N_HEADS) * HEAD_DIM);
  float x1 = base[d];
  float x2 = base[d + 64];
  base[d]      = x1 * c - x2 * s;
  base[d + 64] = x2 * c + x1 * s;
}

// -------------------------------------------------- flash attention (GQA)
// One wave per (16-query block, head). Segments are 16-aligned.
__global__ __launch_bounds__(32) void attn_kernel(
    const float* __restrict__ qkv, const int* __restrict__ cu,
    unsigned short* __restrict__ attn_out) {
  const int q0   = blockIdx.x * 16;
  const int h    = blockIdx.y;
  const int kvh  = h >> 2;
  const int lane = threadIdx.x;
  const int r    = lane & 15;
  const int half = lane >> 4;
  const float scale = 0.08838834764831845f;   // 1/sqrt(128)

  int seg = 0;
#pragma unroll
  for (int i = 0; i < 4; ++i) if (q0 >= cu[i + 1]) seg = i + 1;
  const int s0 = cu[seg];
  const int s1 = cu[seg + 1];

  __shared__ __align__(16) unsigned short Ks[32 * 136];
  __shared__ __align__(16) unsigned short Vs[32 * 136];
  __shared__ __align__(16) unsigned short Ps[16 * 40];

  // Q fragments: 4 chunks of K=32 over head_dim
  Frag16 fq[4];
  const float* qptr = qkv + (size_t)(q0 + r) * QKV_LD + h * HEAD_DIM;
#pragma unroll
  for (int kc = 0; kc < 4; ++kc) {
    fq[kc].q[0] = pack_bf16x8_f(qptr + kc * 32 + half * 8);
    fq[kc].q[1] = pack_bf16x8_f(qptr + kc * 32 + 16 + half * 8);
  }

  v8f o[8];
  float mrow[8], lrow[8];
#pragma unroll
  for (int ot = 0; ot < 8; ++ot)
#pragma unroll
    for (int j = 0; j < 8; ++j) o[ot][j] = 0.0f;
#pragma unroll
  for (int j = 0; j < 8; ++j) { mrow[j] = -3.0e38f; lrow[j] = 0.0f; }

  const int kend = q0 + 16;                 // causal upper bound (exclusive)
  for (int kb = s0; kb < kend; kb += 32) {
    __syncthreads();
    // stage K,V (32 keys x 128 dims), coalesced across lanes, fp32->bf16
    for (int kk = 0; kk < 32; ++kk) {
      int key = kb + kk;
      float4 k4 = {0.f, 0.f, 0.f, 0.f}, v4 = {0.f, 0.f, 0.f, 0.f};
      if (key < s1) {
        const float* kp = qkv + (size_t)key * QKV_LD + D_MODEL + kvh * HEAD_DIM;
        k4 = *(const float4*)(kp + lane * 4);
        v4 = *(const float4*)(kp + KV_HEADS * HEAD_DIM + lane * 4);
      }
      unsigned short* kd = &Ks[kk * 136 + lane * 4];
      kd[0] = f32_to_bf16(k4.x); kd[1] = f32_to_bf16(k4.y);
      kd[2] = f32_to_bf16(k4.z); kd[3] = f32_to_bf16(k4.w);
      unsigned short* vd = &Vs[kk * 136 + lane * 4];
      vd[0] = f32_to_bf16(v4.x); vd[1] = f32_to_bf16(v4.y);
      vd[2] = f32_to_bf16(v4.z); vd[3] = f32_to_bf16(v4.w);
    }
    __syncthreads();

    // S = Q K^T : 16 x 32 (two 16x16 tiles), K-dim = head_dim in 4 chunks
    v8f sacc[2];
#pragma unroll
    for (int nt = 0; nt < 2; ++nt) {
#pragma unroll
      for (int j = 0; j < 8; ++j) sacc[nt][j] = 0.0f;
#pragma unroll
      for (int kc = 0; kc < 4; ++kc) {
        Frag16 fk;
        int key = nt * 16 + r;
        fk.q[0] = *(const uint4*)&Ks[key * 136 + kc * 32 + half * 8];
        fk.q[1] = *(const uint4*)&Ks[key * 136 + kc * 32 + 16 + half * 8];
        sacc[nt] = __builtin_amdgcn_wmma_f32_16x16x32_bf16(
            false, fq[kc].v, false, fk.v, (short)0, sacc[nt], false, false);
      }
    }

    // online softmax; C layout: row = j + 8*half, col = r
#pragma unroll
    for (int j = 0; j < 8; ++j) {
      int qtok = q0 + j + 8 * half;
      int key0 = kb + r, key1 = kb + 16 + r;
      bool ok0 = (key0 <= qtok) && (key0 < s1);
      bool ok1 = (key1 <= qtok) && (key1 < s1);
      float sv0 = ok0 ? sacc[0][j] * scale : -3.0e38f;
      float sv1 = ok1 ? sacc[1][j] * scale : -3.0e38f;
      float mx = fmaxf(sv0, sv1);
#pragma unroll
      for (int off = 1; off < 16; off <<= 1) mx = fmaxf(mx, __shfl_xor(mx, off, 32));
      float mnew = fmaxf(mrow[j], mx);
      float p0 = ok0 ? __expf(sv0 - mnew) : 0.0f;
      float p1 = ok1 ? __expf(sv1 - mnew) : 0.0f;
      float ps = p0 + p1;
#pragma unroll
      for (int off = 1; off < 16; off <<= 1) ps += __shfl_xor(ps, off, 32);
      float alpha = __expf(mrow[j] - mnew);
      mrow[j] = mnew;
      lrow[j] = lrow[j] * alpha + ps;
#pragma unroll
      for (int ot = 0; ot < 8; ++ot) o[ot][j] *= alpha;
      Ps[(j + 8 * half) * 40 + r]      = f32_to_bf16(p0);
      Ps[(j + 8 * half) * 40 + 16 + r] = f32_to_bf16(p1);
    }
    __syncthreads();

    // O += P V : A = P (16x32), B = V (32 keys x 128 dims), 8 n-tiles
    Frag16 fp;
    fp.q[0] = *(const uint4*)&Ps[r * 40 + half * 8];
    fp.q[1] = *(const uint4*)&Ps[r * 40 + 16 + half * 8];
#pragma unroll
    for (int ot = 0; ot < 8; ++ot) {
      unsigned short tv[16];
#pragma unroll
      for (int i = 0; i < 8; ++i) tv[i]     = Vs[(half * 8 + i) * 136 + ot * 16 + r];
#pragma unroll
      for (int i = 0; i < 8; ++i) tv[8 + i] = Vs[(16 + half * 8 + i) * 136 + ot * 16 + r];
      Frag16 fv;
      fv.q[0] = pack_u16x8(tv);
      fv.q[1] = pack_u16x8(tv + 8);
      o[ot] = __builtin_amdgcn_wmma_f32_16x16x32_bf16(
          false, fp.v, false, fv.v, (short)0, o[ot], false, false);
    }
  }

  // normalize and store bf16 (feeds the bf16 O-proj GEMM)
#pragma unroll
  for (int ot = 0; ot < 8; ++ot)
#pragma unroll
    for (int j = 0; j < 8; ++j) {
      float val = o[ot][j] / fmaxf(lrow[j], 1e-20f);
      int m = q0 + j + 8 * half;
      int n = h * HEAD_DIM + ot * 16 + r;
      attn_out[(size_t)m * D_MODEL + n] = f32_to_bf16(val);
    }
}

// ------------------------------------------------------------------ driver
extern "C" void kernel_launch(void* const* d_in, const int* in_sizes, int n_in,
                              void* d_out, int out_size, void* d_ws, size_t ws_size,
                              hipStream_t stream) {
  (void)in_sizes; (void)n_in; (void)out_size; (void)ws_size;
  const float* hidden = (const float*)d_in[0];
  const float* w_qkv  = (const float*)d_in[1];
  const float* w_o    = (const float*)d_in[2];
  const float* cosb   = (const float*)d_in[3];
  const float* sinb   = (const float*)d_in[4];
  const int*   cu     = (const int*)d_in[5];

  char* ws = (char*)d_ws;
  size_t off = 0;
  unsigned short* hs_bf   = (unsigned short*)(ws + off); off += (size_t)T_TOK * D_MODEL * 2;
  unsigned short* wqkv_bf = (unsigned short*)(ws + off); off += (size_t)QKV_N * D_MODEL * 2;
  unsigned short* wo_bf   = (unsigned short*)(ws + off); off += (size_t)D_MODEL * D_MODEL * 2;
  float*          qkv     = (float*)(ws + off);          off += (size_t)T_TOK * QKV_N * 4;
  unsigned short* attn_bf = (unsigned short*)(ws + off); off += (size_t)T_TOK * D_MODEL * 2;

  int n1 = T_TOK * D_MODEL;
  cvt_bf16_kernel<<<(n1 + 255) / 256, 256, 0, stream>>>(hidden, hs_bf, n1);
  int n2 = QKV_N * D_MODEL;
  cvt_bf16_kernel<<<(n2 + 255) / 256, 256, 0, stream>>>(w_qkv, wqkv_bf, n2);
  int n3 = D_MODEL * D_MODEL;
  cvt_bf16_kernel<<<(n3 + 255) / 256, 256, 0, stream>>>(w_o, wo_bf, n3);

  // qkv = clip(hidden @ w_qkv^T, -8, 8)
  gemm_nt_bf16_kernel<<<dim3(QKV_N / 128, T_TOK / 128), 256, 0, stream>>>(
      hs_bf, wqkv_bf, qkv, T_TOK, QKV_N, D_MODEL, 8.0f);
  // RoPE on q and k heads in place
  rope_kernel<<<dim3(T_TOK, N_HEADS + KV_HEADS), 64, 0, stream>>>(qkv, cosb, sinb);
  // varlen causal GQA attention -> bf16 activations
  attn_kernel<<<dim3(T_TOK / 16, N_HEADS), 32, 0, stream>>>(qkv, cu, attn_bf);
  // out = attn @ w_o^T
  gemm_nt_bf16_kernel<<<dim3(D_MODEL / 128, T_TOK / 128), 256, 0, stream>>>(
      attn_bf, wo_bf, (float*)d_out, T_TOK, D_MODEL, D_MODEL, 0.0f);
}
